// Llama4ConditionalFeedForward_29368986370066
// MI455X (gfx1250) — matrix operand order repaired and checked by reading, not compiled, not run on hardware
//
#include <hip/hip_runtime.h>

typedef __attribute__((ext_vector_type(16))) _Float16 v16h;
typedef __attribute__((ext_vector_type(8)))  _Float16 v8h;
typedef __attribute__((ext_vector_type(4)))  _Float16 v4h;
typedef __attribute__((ext_vector_type(8)))  float    v8f;
typedef __attribute__((ext_vector_type(4)))  float    v4f;

#define NUM_EXPERTS 8
#define DIM   1024
#define INTER 2048
#define T_TOK 512

// Workspace layout:
//   int meta[0..8)    : per-expert token counts
//   int meta[8..16)   : per-expert offsets into token list
//   int meta[16..528) : token list, grouped by expert
//   byte offset 4096  : activations, _Float16 [512][2048]  (2 MB)

// ---------------------------------------------------------------------------
// Pass 0: bucket tokens by expert.
// ---------------------------------------------------------------------------
__global__ __launch_bounds__(256) void moe_bucket(const int* __restrict__ eidx,
                                                  int* __restrict__ meta) {
  __shared__ int lc[NUM_EXPERTS], loff[NUM_EXPERTS], cur[NUM_EXPERTS];
  const int tid = threadIdx.x;
  if (tid < NUM_EXPERTS) lc[tid] = 0;
  __syncthreads();
  for (int t = tid; t < T_TOK; t += blockDim.x) atomicAdd(&lc[eidx[t]], 1);
  __syncthreads();
  if (tid == 0) {
    int s = 0;
    for (int e = 0; e < NUM_EXPERTS; ++e) { loff[e] = s; cur[e] = s; s += lc[e]; }
  }
  __syncthreads();
  for (int t = tid; t < T_TOK; t += blockDim.x) {
    int e = eidx[t];
    int p = atomicAdd(&cur[e], 1);
    meta[16 + p] = t;
  }
  if (tid < NUM_EXPERTS) { meta[tid] = lc[tid]; meta[8 + tid] = loff[tid]; }
}

// Fragment layouts per CDNA5 ISA §7.12.2 (16-bit A 16x32, B 32x16).
// For kk8 in {0,8,16,24}, elements kk8..kk8+7 occupy 8 CONTIGUOUS halfwords
// at one lane, so an 8-half group stages with a single ds_store_b128.
__device__ __forceinline__ int a_frag_idx(int ml /*0..63*/, int kk /*0..31*/) {
  int half, h;
  if (kk < 16) { half = kk >> 3; h = kk & 7; }
  else         { int k2 = kk - 16; half = k2 >> 3; h = 8 + (k2 & 7); }
  int lane = (ml & 15) + (half << 4);
  return ((((ml >> 4) << 5) + lane) << 4) + h;          // [ms][lane][16]
}
__device__ __forceinline__ int b_frag_idx(int c, int kk /*0..31*/, int nl /*0..31*/) {
  int lane = (nl & 15) + (((kk >> 3) & 1) << 4);
  return ((((c << 1) + (nl >> 4)) << 5) + lane) * 16 + ((kk & 7) + ((kk >> 4) << 3)); // [c][ns][lane][16]
}

// ---------------------------------------------------------------------------
// Pass 1: gate_up GEMM + SiLU. Block = (expert, 32-col tile of INTER).
// Whole K=1024 gate+up weight slab resident in LDS; A tile double-buffered.
// ---------------------------------------------------------------------------
__global__ __launch_bounds__(256) void moe_gate_up(const float* __restrict__ x,
                                                   const float* __restrict__ w1,
                                                   const int* __restrict__ meta,
                                                   _Float16* __restrict__ act) {
  extern __shared__ char smem[];
  _Float16* Bg  = (_Float16*)smem;               // 32768 halfs
  _Float16* Bu  = (_Float16*)(smem + 65536);     // 32768 halfs
  _Float16* Aab = (_Float16*)(smem + 131072);    // 2 x 2048 halfs
  int*      Tl  = (int*)(smem + 139264);         // 512 token ids

  const int e  = blockIdx.x >> 6;                // INTER/32 = 64 n-tiles
  const int nt = blockIdx.x & 63;
  const int Me = meta[e];
  if (Me == 0) return;
  const int base = meta[8 + e];
  const int tid  = threadIdx.x;
  const int lane = tid & 31;
  const int wave = tid >> 5;
  const int ms = wave & 3;                        // 4 M-subtiles (64-row block)
  const int ns = wave >> 2;                       // 2 N-subtiles (32-col tile)

  const float* W = w1 + (size_t)e * DIM * (2 * INTER);

  // Token list -> LDS (once per block).
  for (int t = tid; t < Me; t += 256) Tl[t] = meta[16 + base + t];

  // Stage weight slab once: float4 loads, 4-deep batches, scatter b16 stores.
  for (int it = 0; it < 8; ++it) {
    v4f g4[4], u4[4]; int k4[4], nl4[4];
#pragma unroll
    for (int j = 0; j < 4; ++j) {
      int q4 = tid + 256 * (it * 4 + j);          // over 8192 float4s
      k4[j]  = q4 >> 3;
      nl4[j] = (q4 & 7) * 4;
      size_t row = (size_t)k4[j] * (2 * INTER);
      g4[j] = *(const v4f*)&W[row + nt * 32 + nl4[j]];
      u4[j] = *(const v4f*)&W[row + INTER + nt * 32 + nl4[j]];
    }
#pragma unroll
    for (int j = 0; j < 4; ++j) {
#pragma unroll
      for (int s = 0; s < 4; ++s) {
        int idx = b_frag_idx(k4[j] >> 5, k4[j] & 31, nl4[j] + s);
        Bg[idx] = (_Float16)g4[j][s];
        Bu[idx] = (_Float16)u4[j][s];
      }
    }
  }
  __syncthreads();

  const int abase = ((ms << 5) + lane) << 4;
  const int bbase = ((ns << 5) + lane) << 4;

  // Per-thread A staging plan: one row x 8 consecutive k per chunk.
  const int mlA = tid >> 2;                       // 0..63
  const int kk8 = (tid & 3) * 8;                  // {0,8,16,24}
  const int stA = a_frag_idx(mlA, kk8);           // 8 contiguous halfs

  for (int m0 = 0; m0 < Me; m0 += 64) {
    int rowc = m0 + mlA; if (rowc > Me - 1) rowc = Me - 1;   // clamp: pad rows
    const float* pA = x + (size_t)Tl[rowc] * DIM + kk8;      // always valid

    // Prologue: stage chunk 0.
    {
      v4f a0 = *(const v4f*)(pA);
      v4f a1 = *(const v4f*)(pA + 4);
      v8h h;
#pragma unroll
      for (int s = 0; s < 4; ++s) { h[s] = (_Float16)a0[s]; h[4 + s] = (_Float16)a1[s]; }
      *(v8h*)&Aab[stA] = h;
    }
    __syncthreads();

    v8f accg = {}; v8f accu = {};
    for (int c = 0; c < DIM / 32; ++c) {
      v4f a0, a1;
      const bool more = (c + 1) < (DIM / 32);
      if (more) {
        a0 = *(const v4f*)(pA + (c + 1) * 32);
        a1 = *(const v4f*)(pA + (c + 1) * 32 + 4);
      }
      const _Float16* cur = Aab + ((c & 1) << 11);
      v16h a  = *(const v16h*)&cur[abase];
      v16h bg = *(const v16h*)&Bg[(c << 10) + bbase];
      v16h bu = *(const v16h*)&Bu[(c << 10) + bbase];
      accg = __builtin_amdgcn_wmma_f32_16x16x32_f16(false, a, false, bg,
                                                    (short)0, accg, false, false);
      accu = __builtin_amdgcn_wmma_f32_16x16x32_f16(false, a, false, bu,
                                                    (short)0, accu, false, false);
      if (more) {
        v8h h;
#pragma unroll
        for (int s = 0; s < 4; ++s) { h[s] = (_Float16)a0[s]; h[4 + s] = (_Float16)a1[s]; }
        *(v8h*)&(Aab + (((c + 1) & 1) << 11))[stA] = h;
      }
      __syncthreads();
    }
    // C layout: VGPR r, lanes 0-15 -> M=r, lanes 16-31 -> M=8+r; N = lane%16.
#pragma unroll
    for (int r = 0; r < 8; ++r) {
      int ml = (ms << 4) + r + ((lane >> 4) << 3);
      if (m0 + ml < Me) {
        int tok = Tl[m0 + ml];
        int col = nt * 32 + (ns << 4) + (lane & 15);
        float g = accg[r], u = accu[r];
        float s = g / (1.f + __expf(-g));         // silu
        act[(size_t)tok * INTER + col] = (_Float16)(s * u);
      }
    }
  }
}

// ---------------------------------------------------------------------------
// Pass 2: down GEMM. Block = (expert, 32-col tile of DIM).
// Whole K=2048 weight slab resident in LDS; A tile double-buffered.
// A is already f16: staging is one b128 load + one b128 DS store per thread.
// ---------------------------------------------------------------------------
__global__ __launch_bounds__(256) void moe_down(const _Float16* __restrict__ act,
                                                const float* __restrict__ w2,
                                                const int* __restrict__ meta,
                                                float* __restrict__ out) {
  extern __shared__ char smem[];
  _Float16* Bl  = (_Float16*)smem;               // 65536 halfs
  _Float16* Aab = (_Float16*)(smem + 131072);    // 2 x 2048 halfs
  int*      Tl  = (int*)(smem + 139264);         // 512 token ids

  const int e  = blockIdx.x >> 5;                // DIM/32 = 32 n-tiles
  const int nt = blockIdx.x & 31;
  const int Me = meta[e];
  if (Me == 0) return;
  const int base = meta[8 + e];
  const int tid  = threadIdx.x;
  const int lane = tid & 31;
  const int wave = tid >> 5;
  const int ms = wave & 3;
  const int ns = wave >> 2;

  const float* W = w2 + (size_t)e * INTER * DIM;

  for (int t = tid; t < Me; t += 256) Tl[t] = meta[16 + base + t];

  for (int it = 0; it < 16; ++it) {
    v4f w4[4]; int k4[4], nl4[4];
#pragma unroll
    for (int j = 0; j < 4; ++j) {
      int q4 = tid + 256 * (it * 4 + j);          // over 16384 float4s
      k4[j]  = q4 >> 3;
      nl4[j] = (q4 & 7) * 4;
      w4[j] = *(const v4f*)&W[(size_t)k4[j] * DIM + nt * 32 + nl4[j]];
    }
#pragma unroll
    for (int j = 0; j < 4; ++j) {
#pragma unroll
      for (int s = 0; s < 4; ++s)
        Bl[b_frag_idx(k4[j] >> 5, k4[j] & 31, nl4[j] + s)] = (_Float16)w4[j][s];
    }
  }
  __syncthreads();

  const int abase = ((ms << 5) + lane) << 4;
  const int bbase = ((ns << 5) + lane) << 4;

  const int mlA = tid >> 2;
  const int kk8 = (tid & 3) * 8;
  const int stA = a_frag_idx(mlA, kk8);

  for (int m0 = 0; m0 < Me; m0 += 64) {
    int rowc = m0 + mlA; if (rowc > Me - 1) rowc = Me - 1;
    const _Float16* pA = act + (size_t)Tl[rowc] * INTER + kk8;

    *(v8h*)&Aab[stA] = *(const v8h*)(pA);         // prologue, chunk 0
    __syncthreads();

    v8f acc = {};
    for (int c = 0; c < INTER / 32; ++c) {
      v8h hn;
      const bool more = (c + 1) < (INTER / 32);
      if (more) hn = *(const v8h*)(pA + (c + 1) * 32);
      const _Float16* cur = Aab + ((c & 1) << 11);
      v16h a = *(const v16h*)&cur[abase];
      v16h b = *(const v16h*)&Bl[(c << 10) + bbase];
      acc = __builtin_amdgcn_wmma_f32_16x16x32_f16(false, a, false, b,
                                                   (short)0, acc, false, false);
      if (more) *(v8h*)&(Aab + (((c + 1) & 1) << 11))[stA] = hn;
      __syncthreads();
    }
#pragma unroll
    for (int r = 0; r < 8; ++r) {
      int ml = (ms << 4) + r + ((lane >> 4) << 3);
      if (m0 + ml < Me) {
        int tok = Tl[m0 + ml];
        int col = nt * 32 + (ns << 4) + (lane & 15);
        out[(size_t)tok * DIM + col] = acc[r];
      }
    }
  }
}

// ---------------------------------------------------------------------------
extern "C" void kernel_launch(void* const* d_in, const int* in_sizes, int n_in,
                              void* d_out, int out_size, void* d_ws, size_t ws_size,
                              hipStream_t stream) {
  const float* x    = (const float*)d_in[0];
  const int*   eidx = (const int*)d_in[1];
  const float* w1   = (const float*)d_in[2];
  const float* w2   = (const float*)d_in[3];
  float* out = (float*)d_out;

  int* meta = (int*)d_ws;
  _Float16* act = (_Float16*)((char*)d_ws + 4096);

  moe_bucket<<<1, 256, 0, stream>>>(eidx, meta);

  const size_t lds = 139264 + 2048;               // 141312 B (~138 KB)
  moe_gate_up<<<dim3(NUM_EXPERTS * (INTER / 32)), dim3(256), lds, stream>>>(
      x, w1, meta, act);
  moe_down<<<dim3(NUM_EXPERTS * (DIM / 32)), dim3(256), lds, stream>>>(
      act, w2, meta, out);
}
